// Model_69612829934235
// MI455X (gfx1250) — compile-verified, matrix operand order checked by reading, not run on hardware
//
#include <hip/hip_runtime.h>
#include <math.h>

typedef __attribute__((ext_vector_type(16))) _Float16 v16h;
typedef __attribute__((ext_vector_type(8)))  _Float16 v8h;
typedef __attribute__((ext_vector_type(4)))  _Float16 v4h;
typedef __attribute__((ext_vector_type(8)))  float    v8f;
typedef __attribute__((ext_vector_type(4)))  float    v4f;

#define ND 8000
#define NL 16000
#define NE 60000
#define NM 50000
#define NH 16

#define BM 128
#define BN 64
#define BK 32
#define SROW 40   // halves per LDS row: 80 B, 16-B aligned, bank-spreading

// ---------------------------------------------------------------- utilities
__global__ void fill_f32(float* __restrict__ p, float v, int n) {
  int i = blockIdx.x * blockDim.x + threadIdx.x;
  if (i < n) p[i] = v;
}

// out[n,c] = emb[id[n]*C + c]   (C multiple of 4)
__global__ void gather_rows(const float* __restrict__ emb, const int* __restrict__ id,
                            float* __restrict__ out, int n, int c) {
  int i = blockIdx.x * blockDim.x + threadIdx.x;
  int q = c >> 2;
  if (i >= n * q) return;
  int r = i / q, cc = (i - r * q) << 2;
  *(v4f*)&out[(size_t)r * c + cc] = *(const v4f*)&emb[(size_t)id[r] * c + cc];
}

// ------------------------------------------------------- WMMA GEMM (f16 in, f32 acc)
// C[M,N] (+)= A[M,K] @ B[K,N] (+bias) (relu). Row-major f32 I/O.
// Block = 256 threads = 8 waves; tile 128(M) x 64(N); K-step 32.
// Each wave owns a 32x32 subtile = 2x2 v_wmma_f32_16x16x32_f16 accumulators.
// B is staged TRANSPOSED in LDS so both A and B fragments are two contiguous
// 16-byte runs -> ds_load_b128 pairs (ISA 7.12.2 16-bit operand layout).
__global__ __launch_bounds__(256)
void gemm_wmma_f16(const float* __restrict__ A, const float* __restrict__ B,
                   float* __restrict__ C, int M, int N, int K,
                   const float* __restrict__ bias, int accumulate, int relu) {
  __shared__ _Float16 sA[BM][SROW];    // [m][k]
  __shared__ _Float16 sBt[BN][SROW];   // [n][k]  (transposed)

  const int tid    = threadIdx.x;
  const int wave   = tid >> 5;
  const int lane   = tid & 31;
  const int row16  = lane & 15;
  const int laneHi = lane >> 4;        // 0: lanes 0-15, 1: lanes 16-31
  const int waveM  = wave & 3;         // 4 M-slots of 32 rows
  const int waveN  = wave >> 2;        // 2 N-slots of 32 cols
  const int m0 = blockIdx.y * BM;
  const int n0 = blockIdx.x * BN;

  v8f acc[2][2] = {};

  for (int k0 = 0; k0 < K; k0 += BK) {
    // ---- stage A tile: 128x32, 16 elems/thread (4 x b128 global, 4 x b64 LDS)
    {
      int r = tid >> 1;
      int cb = (tid & 1) * 16;
      int gm = m0 + r;
      if (gm < M) {
        const float* Ap = A + (size_t)gm * K + k0 + cb;
        if (k0 + cb + 16 <= K) {
#pragma unroll
          for (int q = 0; q < 4; ++q) {
            v4f x = *(const v4f*)(Ap + 4 * q);
            v4h h; h[0] = (_Float16)x[0]; h[1] = (_Float16)x[1];
                   h[2] = (_Float16)x[2]; h[3] = (_Float16)x[3];
            *(v4h*)&sA[r][cb + 4 * q] = h;
          }
        } else {
#pragma unroll
          for (int i = 0; i < 16; ++i) {
            int gk = k0 + cb + i;
            sA[r][cb + i] = (_Float16)((gk < K) ? Ap[i] : 0.0f);
          }
        }
      } else {
        v4h z = {};
#pragma unroll
        for (int q = 0; q < 4; ++q) *(v4h*)&sA[r][cb + 4 * q] = z;
      }
    }
    // ---- stage B tile transposed: 32(k) x 64(n), 8 elems/thread
    {
      int r = tid >> 3;              // k row within tile
      int cb = (tid & 7) * 8;        // n col base
      int gk = k0 + r;
      if (gk < K && n0 + cb + 8 <= N) {
        const float* Bp = B + (size_t)gk * N + n0 + cb;
#pragma unroll
        for (int q = 0; q < 2; ++q) {
          v4f x = *(const v4f*)(Bp + 4 * q);
#pragma unroll
          for (int i = 0; i < 4; ++i)
            sBt[cb + 4 * q + i][r] = (_Float16)x[i];
        }
      } else {
#pragma unroll
        for (int i = 0; i < 8; ++i) {
          int gn = n0 + cb + i;
          float v = (gk < K && gn < N) ? B[(size_t)gk * N + gn] : 0.0f;
          sBt[cb + i][r] = (_Float16)v;
        }
      }
    }
    __syncthreads();

    // ---- fragments: two contiguous 16-B runs each (ds_load_b128 pairs)
    v16h af[2], bfr[2];
#pragma unroll
    for (int t = 0; t < 2; ++t) {
      int ar = waveM * 32 + t * 16 + row16;   // A row (M)
      int bc = waveN * 32 + t * 16 + row16;   // B col (N)
      v8h alo = *(const v8h*)&sA[ar][laneHi * 8];
      v8h ahi = *(const v8h*)&sA[ar][16 + laneHi * 8];
      v8h blo = *(const v8h*)&sBt[bc][laneHi * 8];
      v8h bhi = *(const v8h*)&sBt[bc][16 + laneHi * 8];
#pragma unroll
      for (int i = 0; i < 8; ++i) {
        af[t][i] = alo[i];  af[t][8 + i] = ahi[i];
        bfr[t][i] = blo[i]; bfr[t][8 + i] = bhi[i];
      }
    }
#pragma unroll
    for (int tm = 0; tm < 2; ++tm)
#pragma unroll
      for (int tn = 0; tn < 2; ++tn)
        acc[tm][tn] = __builtin_amdgcn_wmma_f32_16x16x32_f16(
            false, af[tm], false, bfr[tn], (short)0, acc[tm][tn], false, false);
    __syncthreads();
  }

  // ---- epilogue. C/D layout: VGPR v -> M = v + 8*laneHi, N = lane%16
#pragma unroll
  for (int tm = 0; tm < 2; ++tm) {
#pragma unroll
    for (int tn = 0; tn < 2; ++tn) {
      int gn = n0 + waveN * 32 + tn * 16 + row16;
#pragma unroll
      for (int v = 0; v < 8; ++v) {
        int gm = m0 + waveM * 32 + tm * 16 + v + laneHi * 8;
        if (gm < M && gn < N) {
          size_t idx = (size_t)gm * N + gn;
          float val = acc[tm][tn][v];
          if (bias) val += bias[gn];
          if (accumulate) val += C[idx];
          if (relu && val < 0.0f) val = 0.0f;
          C[idx] = val;
        }
      }
    }
  }
}

// -------------------------------------------------- attention score machinery
// Wa[k*NH+h] = sum_c W[k, h*Cc+c] * a[h,c]   (folds zd out of existence)
__global__ void attn_proj(const float* __restrict__ W, const float* __restrict__ a,
                          float* __restrict__ Wa, int Cc) {
  int i = blockIdx.x * blockDim.x + threadIdx.x;
  if (i >= 128 * NH) return;
  int k = i / NH, h = i - k * NH;
  float s = 0.0f;
  for (int c = 0; c < Cc; ++c) s += W[(size_t)k * (NH * Cc) + h * Cc + c] * a[h * Cc + c];
  Wa[i] = s;
}

// out[n*NH+h] = sum_k h[n,k] * Wa[k*NH+h]  (K = 128)
__global__ void node_scores(const float* __restrict__ hmat, const float* __restrict__ Wa,
                            float* __restrict__ out, int n) {
  int i = blockIdx.x * blockDim.x + threadIdx.x;
  if (i >= n * NH) return;
  int r = i / NH, h = i - r * NH;
  float s = 0.0f;
  for (int k = 0; k < 128; ++k) s += hmat[(size_t)r * 128 + k] * Wa[k * NH + h];
  out[i] = s;
}

__device__ __forceinline__ void atomicMaxF(float* addr, float value) {
  if (value >= 0.0f) atomicMax((int*)addr, __float_as_int(value));
  else               atomicMin((unsigned int*)addr, __float_as_uint(value));
}

__global__ void edge_max(const float* __restrict__ szs, const float* __restrict__ szd,
                         const int* __restrict__ src, const int* __restrict__ dst,
                         float* __restrict__ mmax, int ne) {
  int i = blockIdx.x * blockDim.x + threadIdx.x;
  if (i >= ne * NH) return;
  int e = i / NH, h = i - e * NH;
  float al = szs[src[e] * NH + h] + szd[dst[e] * NH + h];
  al = (al > 0.0f) ? al : 0.2f * al;                 // leaky_relu(0.2)
  atomicMaxF(&mmax[dst[e] * NH + h], al);
}

__global__ void edge_exp(const float* __restrict__ szs, const float* __restrict__ szd,
                         const int* __restrict__ src, const int* __restrict__ dst,
                         const float* __restrict__ mmax, float* __restrict__ eexp,
                         float* __restrict__ ssum, int ne) {
  int i = blockIdx.x * blockDim.x + threadIdx.x;
  if (i >= ne * NH) return;
  int e = i / NH, h = i - e * NH;
  int d = dst[e];
  float al = szs[src[e] * NH + h] + szd[d * NH + h];
  al = (al > 0.0f) ? al : 0.2f * al;
  float mv = mmax[d * NH + h];
  if (!(mv > -3.0e38f)) mv = 0.0f;                   // finite guard
  float ex = __expf(al - mv);
  eexp[i] = ex;
  atomicAdd(&ssum[d * NH + h], ex);
}

// one block per edge: agg[dst, h*Cc+c] += zs[src, h*Cc+c] * alpha[h]
__global__ __launch_bounds__(256)
void edge_aggregate(const float* __restrict__ zs, const float* __restrict__ eexp,
                    const float* __restrict__ ssum, const int* __restrict__ src,
                    const int* __restrict__ dst, float* __restrict__ agg, int Cc) {
  __shared__ float salpha[NH];
  int e = blockIdx.x;
  int s = src[e], d = dst[e];
  if (threadIdx.x < NH)
    salpha[threadIdx.x] = eexp[e * NH + threadIdx.x] / (ssum[d * NH + threadIdx.x] + 1e-16f);
  __syncthreads();
  int HC = NH * Cc;
  for (int c = threadIdx.x; c < HC; c += 256) {
    int h = c / Cc;
    atomicAdd(&agg[(size_t)d * HC + c], zs[(size_t)s * HC + c] * salpha[h]);
  }
}

// out[n,c] = mean_h relu(agg[n, h*Cc+c] + b[h*Cc+c])
__global__ void head_mean(const float* __restrict__ agg, const float* __restrict__ b,
                          float* __restrict__ out, int n, int Cc) {
  int i = blockIdx.x * blockDim.x + threadIdx.x;
  if (i >= n * Cc) return;
  int r = i / Cc, c = i - r * Cc;
  float s = 0.0f;
  for (int h = 0; h < NH; ++h) {
    float v = agg[(size_t)r * NH * Cc + h * Cc + c] + b[h * Cc + c];
    s += (v > 0.0f) ? v : 0.0f;
  }
  out[i] = s * (1.0f / NH);
}

// h[n,:] *= exp(h[n,:] @ pw + pb)   (C = 32)
__global__ void penalty_scale(float* __restrict__ hmat, const float* __restrict__ pw,
                              const float* __restrict__ pb, int n) {
  int i = blockIdx.x * blockDim.x + threadIdx.x;
  if (i >= n) return;
  float s = 0.0f;
  for (int k = 0; k < 32; ++k) s += hmat[(size_t)i * 32 + k] * pw[k];
  float g = __expf(s + pb[0]);
  for (int k = 0; k < 32; ++k) hmat[(size_t)i * 32 + k] *= g;
}

// f[m, 0:32] = hd[ls[m]], f[m, 32:64] = hl[ld[m]]  (vectorized: 16 x v4f per row)
__global__ void gather_pairs(const float* __restrict__ hd, const float* __restrict__ hl,
                             const int* __restrict__ ls, const int* __restrict__ ld,
                             float* __restrict__ f) {
  int i = blockIdx.x * blockDim.x + threadIdx.x;
  if (i >= NM * 16) return;
  int m = i >> 4, q = (i & 15) << 2;   // q: 0..60 step 4
  v4f x = (q < 32) ? *(const v4f*)&hd[(size_t)ls[m] * 32 + q]
                   : *(const v4f*)&hl[(size_t)ld[m] * 32 + (q - 32)];
  *(v4f*)&f[(size_t)m * 64 + q] = x;
}

__global__ void final_fc2(const float* __restrict__ g1, const float* __restrict__ fc2w,
                          const float* __restrict__ fc2b, float* __restrict__ out) {
  int m = blockIdx.x * blockDim.x + threadIdx.x;
  if (m >= NM) return;
  float s = 0.0f;
  for (int k = 0; k < 64; ++k) s += g1[(size_t)m * 64 + k] * fc2w[k];
  out[m] = s + fc2b[0];
}

// ------------------------------------------------------------- host wiring
static void run_gemm(const float* A, const float* B, float* C, int M, int N, int K,
                     const float* bias, int acc, int relu, hipStream_t s) {
  dim3 g((N + BN - 1) / BN, (M + BM - 1) / BM);
  gemm_wmma_f16<<<g, 256, 0, s>>>(A, B, C, M, N, K, bias, acc, relu);
}

static void run_dir(const float* hsrc, int nsrc, const float* hdst, int ndst,
                    const int* srcIdx, const int* dstIdx,
                    const float* Ws, const float* Wd, const float* as_, const float* ad_,
                    const float* b, int Cc,
                    float* pool, float* szs, float* szd, float* mmax, float* ssum,
                    float* eexp, float* Wsa, float* Wda, float* out, hipStream_t s) {
  const int HC = NH * Cc;
  float* zs  = pool;
  float* agg = pool + (size_t)nsrc * HC;
  attn_proj<<<(128 * NH + 255) / 256, 256, 0, s>>>(Ws, as_, Wsa, Cc);
  attn_proj<<<(128 * NH + 255) / 256, 256, 0, s>>>(Wd, ad_, Wda, Cc);
  run_gemm(hsrc, Ws, zs, nsrc, HC, 128, nullptr, 0, 0, s);
  node_scores<<<(nsrc * NH + 255) / 256, 256, 0, s>>>(hsrc, Wsa, szs, nsrc);
  node_scores<<<(ndst * NH + 255) / 256, 256, 0, s>>>(hdst, Wda, szd, ndst);
  fill_f32<<<(ndst * NH + 255) / 256, 256, 0, s>>>(mmax, -INFINITY, ndst * NH);
  fill_f32<<<(ndst * NH + 255) / 256, 256, 0, s>>>(ssum, 0.0f, ndst * NH);
  edge_max<<<(NE * NH + 255) / 256, 256, 0, s>>>(szs, szd, srcIdx, dstIdx, mmax, NE);
  edge_exp<<<(NE * NH + 255) / 256, 256, 0, s>>>(szs, szd, srcIdx, dstIdx, mmax, eexp, ssum, NE);
  int aggN = ndst * HC;
  fill_f32<<<(aggN + 255) / 256, 256, 0, s>>>(agg, 0.0f, aggN);
  edge_aggregate<<<NE, 256, 0, s>>>(zs, eexp, ssum, srcIdx, dstIdx, agg, Cc);
  head_mean<<<(ndst * Cc + 255) / 256, 256, 0, s>>>(agg, b, out, ndst, Cc);
}

extern "C" void kernel_launch(void* const* d_in, const int* in_sizes, int n_in,
                              void* d_out, int out_size, void* d_ws, size_t ws_size,
                              hipStream_t stream) {
  (void)in_sizes; (void)n_in; (void)out_size; (void)ws_size;
  const float* x_d       = (const float*)d_in[0];
  const float* x_l       = (const float*)d_in[1];
  const int*   node_id_d = (const int*)d_in[2];
  const int*   node_id_l = (const int*)d_in[3];
  const int*   edge_src  = (const int*)d_in[4];
  const int*   edge_dst  = (const int*)d_in[5];
  const int*   label_src = (const int*)d_in[6];
  const int*   label_dst = (const int*)d_in[7];
  const float* emb_d  = (const float*)d_in[8];
  const float* emb_l  = (const float*)d_in[9];
  const float* lin_dw = (const float*)d_in[10];
  const float* lin_db = (const float*)d_in[11];
  const float* lin_lw = (const float*)d_in[12];
  const float* lin_lb = (const float*)d_in[13];
  const float* Wsrc[3] = {(const float*)d_in[14], (const float*)d_in[19], (const float*)d_in[24]};
  const float* Wdst[3] = {(const float*)d_in[15], (const float*)d_in[20], (const float*)d_in[25]};
  const float* asrc[3] = {(const float*)d_in[16], (const float*)d_in[21], (const float*)d_in[26]};
  const float* adst[3] = {(const float*)d_in[17], (const float*)d_in[22], (const float*)d_in[27]};
  const float* bb[3]   = {(const float*)d_in[18], (const float*)d_in[23], (const float*)d_in[28]};
  const float* pw   = (const float*)d_in[29];
  const float* pb   = (const float*)d_in[30];
  const float* fc1w = (const float*)d_in[31];
  const float* fc1b = (const float*)d_in[32];
  const float* fc2w = (const float*)d_in[33];
  const float* fc2b = (const float*)d_in[34];

  // ---- workspace carve (~229 MB of f32) ----
  float* ws = (float*)d_ws;
  size_t off = 0;
  float* pool = ws + off; off += (size_t)(ND + NL) * (NH * 128);  // 24000 x 2048
  float* hdA  = ws + off; off += (size_t)ND * 128;
  float* hlA  = ws + off; off += (size_t)NL * 128;
  float* hdB  = ws + off; off += (size_t)ND * 128;
  float* hlB  = ws + off; off += (size_t)NL * 128;
  float* szs  = ws + off; off += (size_t)NL * NH;
  float* szd  = ws + off; off += (size_t)NL * NH;
  float* mmax = ws + off; off += (size_t)NL * NH;
  float* ssum = ws + off; off += (size_t)NL * NH;
  float* eexp = ws + off; off += (size_t)NE * NH;
  float* Wsa  = ws + off; off += 128 * NH;
  float* Wda  = ws + off; off += 128 * NH;

  // ---- initial embeddings: h = x @ linW + linB + emb[node_id] ----
  gather_rows<<<(ND * 32 + 255) / 256, 256, 0, stream>>>(emb_d, node_id_d, hdA, ND, 128);
  run_gemm(x_d, lin_dw, hdA, ND, 128, 412, lin_db, /*acc=*/1, /*relu=*/0, stream);
  gather_rows<<<(NL * 32 + 255) / 256, 256, 0, stream>>>(emb_l, node_id_l, hlA, NL, 128);
  run_gemm(x_l, lin_lw, hlA, NL, 128, 240, lin_lb, /*acc=*/1, /*relu=*/0, stream);

  float* hd = hdA; float* hl = hlA; float* hdN = hdB; float* hlN = hlB;
  const int Ccs[3] = {128, 128, 32};
  for (int L = 0; L < 3; ++L) {
    const int Cc = Ccs[L];
    const int HC = NH * Cc;
    // dir 0: disease -> lncrna (target hl), params slice [0]
    run_dir(hd, ND, hl, NL, edge_src, edge_dst,
            Wsrc[L], Wdst[L], asrc[L], adst[L], bb[L], Cc,
            pool, szs, szd, mmax, ssum, eexp, Wsa, Wda, hlN, stream);
    // dir 1: lncrna -> disease (target hd), params slice [1]
    run_dir(hl, NL, hd, ND, edge_dst, edge_src,
            Wsrc[L] + (size_t)128 * HC, Wdst[L] + (size_t)128 * HC,
            asrc[L] + NH * Cc, adst[L] + NH * Cc, bb[L] + HC, Cc,
            pool, szs, szd, mmax, ssum, eexp, Wsa, Wda, hdN, stream);
    float* t;
    t = hd; hd = hdN; hdN = t;
    t = hl; hl = hlN; hlN = t;
  }

  // ---- penalty scaling (node dim now 32) ----
  penalty_scale<<<(ND + 255) / 256, 256, 0, stream>>>(hd, pw, pb, ND);
  penalty_scale<<<(NL + 255) / 256, 256, 0, stream>>>(hl, pw + 32, pb + 1, NL);

  // ---- final MLP on label pairs (pool is free now) ----
  float* f  = pool;
  float* g1 = pool + (size_t)NM * 64;
  gather_pairs<<<(NM * 16 + 255) / 256, 256, 0, stream>>>(hd, hl, label_src, label_dst, f);
  run_gemm(f, fc1w, g1, NM, 64, 64, fc1b, /*acc=*/0, /*relu=*/1, stream);
  final_fc2<<<(NM + 255) / 256, 256, 0, stream>>>(g1, fc2w, fc2b, (float*)d_out);
}
